// MPCPolicy_25640954757882
// MI455X (gfx1250) — matrix operand order repaired and checked by reading, not compiled, not run on hardware
//
#include <hip/hip_runtime.h>
#include <hip/hip_bf16.h>

// ---------------------------------------------------------------------------
// CDNA5 (gfx1250, wave32) WMMA types
// ---------------------------------------------------------------------------
typedef __attribute__((ext_vector_type(16))) _Float16 v16h;
typedef __attribute__((ext_vector_type(8)))  float    v8f;

union AV { v16h v; _Float16 h[16]; unsigned u[8]; };

__device__ inline v8f wmma_f16(v16h a, v16h b, v8f c) {
#if defined(__HIP_DEVICE_COMPILE__)
  // v_wmma_f32_16x16x32_f16  D = A*B + C
  return __builtin_amdgcn_wmma_f32_16x16x32_f16(false, a, false, b, (short)0, c,
                                                false, false);
#else
  return c;
#endif
}

// Tensor Data Mover availability (6-arg amdgpu-toolchain/clang-23 form,
// confirmed by the round-2 diagnostic: "expected 6, have 5")
#if defined(__HIP_DEVICE_COMPILE__) && __has_builtin(__builtin_amdgcn_tensor_load_to_lds)
#define HAVE_TDM 1
#else
#define HAVE_TDM 0
#endif

#if HAVE_TDM
typedef __attribute__((ext_vector_type(4))) unsigned tdm_g0_t;
typedef __attribute__((ext_vector_type(8))) int      tdm_g1_t;
typedef __attribute__((ext_vector_type(4))) int      tdm_g2_t;
typedef __attribute__((ext_vector_type(8))) int      tdm_g4_t;

// Load a [rows x 16] f16 tile (row stride 320 elems) from global into LDS with
// row pitch 96 elems (pad: 8 dwords data, then 40 dwords skip).
__device__ inline void tdm_load_actions(const _Float16* gsrc, unsigned lds_byte_addr,
                                        int rows) {
  unsigned long long ga = (unsigned long long)(const void*)gsrc;
  tdm_g0_t g0;
  g0[0] = 1u;                                   // count=1 (valid descriptor)
  g0[1] = lds_byte_addr;                        // lds_addr
  g0[2] = (unsigned)(ga & 0xffffffffull);       // global_addr[31:0]
  g0[3] = (unsigned)((ga >> 32) & 0x01ffffffull) | (2u << 30);  // addr[56:32]|type=2
  tdm_g1_t g1;
  // data_size=1 (2B) | pad_enable | pad_interval=2 (8 dwords) | pad_amount=39 (40 dw)
  g1[0] = (1 << 16) | (1 << 20) | (2 << 22) | (39 << 25);
  g1[1] = (320 << 16);                          // tensor_dim0 = 320 (low16)
  g1[2] = (4096 << 16);                         // tensor_dim0 hi=0 | tensor_dim1 low16
  g1[3] = (16 << 16);                           // tensor_dim1 hi=0 | tile_dim0 = 16
  g1[4] = rows;                                 // tile_dim1 = rows, tile_dim2 = 0
  g1[5] = 320;                                  // tensor_dim0_stride low32
  g1[6] = 0;                                    // stride hi | dim1_stride low (unused 2D)
  g1[7] = 0;
  tdm_g2_t gz  = {0, 0, 0, 0};
  tdm_g4_t gz8 = {0, 0, 0, 0, 0, 0, 0, 0};
  __builtin_amdgcn_tensor_load_to_lds(g0, g1, gz, gz, gz8, 0);
}
#endif

// ---------------------------------------------------------------------------
// Deterministic hash RNG (stands in for jax PRNG; no correctness oracle here)
// ---------------------------------------------------------------------------
__device__ inline unsigned pcg_hash(unsigned x) {
  x = x * 747796405u + 2891336453u;
  unsigned w = ((x >> ((x >> 28u) + 4u)) ^ x) * 277803737u;
  return (w >> 22u) ^ w;
}
__device__ inline unsigned mix2(unsigned a, unsigned b) {
  return pcg_hash(a ^ (b * 0x9E3779B9u));
}
__device__ inline float u01(unsigned s) {
  return ((pcg_hash(s) >> 8) + 0.5f) * (1.0f / 16777216.0f);
}
// Box-Muller pair: one log+sqrt per two normals
__device__ inline float2 rng_normal2(unsigned s) {
  float u1 = u01(s);
  float u2 = u01(s ^ 0x6E624EB7u);
  float r = sqrtf(-2.0f * logf(u1));
  float a = 6.28318530718f * u2;
  return make_float2(r * cosf(a), r * sinf(a));
}
__device__ inline float rng_normal(unsigned s) {
  float u1 = u01(s);
  float u2 = u01(s ^ 0x6E624EB7u);
  return sqrtf(-2.0f * logf(u1)) * cosf(6.28318530718f * u2);
}
__device__ inline float softplusf(float x) {
  return (x > 20.0f) ? x : log1pf(expf(x));
}
// Pade(7,6) tanh: pure FMA, no transcendentals (WMMA co-executes with VALU,
// so trimming VALU trans ops is the rollout critical path on CDNA5).
__device__ inline float fast_tanh(float x) {
  x = fminf(4.0f, fmaxf(-4.0f, x));
  float x2 = x * x;
  float p = x * (135135.0f + x2 * (17325.0f + x2 * (378.0f + x2)));
  float q = 135135.0f + x2 * (62370.0f + x2 * (3150.0f + x2 * 28.0f));
  return p / q;
}

// ---------------------------------------------------------------------------
// Operand loaders
//   B operand: pre-packed so lane L reads 16 contiguous halves (32B) per tile
//   A operand: LDS row-major [row][ldk], gathered in contiguous k order (same as packed B):
//     lane L: m = L%16 ; khalf = 16*(L/16)
//     vgpr q : k = 2q + khalf (contiguous, matches packed B)
// ---------------------------------------------------------------------------
__device__ inline v16h load_b_g(const _Float16* __restrict__ base, int tile, int lane) {
  return *(const v16h*)(base + (size_t)tile * 512 + lane * 16);
}
__device__ inline v16h load_a_lds(const _Float16* lds, int mbase, int ldk,
                                  int kbase, int lane) {
  AV r;
  const unsigned* row = (const unsigned*)(lds + (size_t)(mbase + (lane & 15)) * ldk);
  int khalf = (lane >> 4) << 4;
#pragma unroll
  for (int q = 0; q < 8; ++q) {
    int k0 = 2 * q + khalf + kbase;
    r.u[q] = row[k0 >> 1];  // k0 always even
  }
  return r.v;
}

// ---------------------------------------------------------------------------
// Problem constants
// ---------------------------------------------------------------------------
#define H_PLAN 20
#define A_DIM 16
#define S_DIM 64
#define N_CAND 4096
#define TOP_K 128
#define HID 512
#define DDIM 320  // H_PLAN * A_DIM

// Workspace layout (byte offsets, all 256B aligned)
static constexpr size_t OFF_W1B  = 0;                              // 96*512 f16
static constexpr size_t OFF_WR1B = OFF_W1B  + 96 * 512 * 2;        // 96*512 f16
static constexpr size_t OFF_W2B  = OFF_WR1B + 96 * 512 * 2;        // 512*128 f16
static constexpr size_t OFF_WR2B = OFF_W2B  + 512 * 128 * 2;       // 512*16 f16
static constexpr size_t OFF_LB   = OFF_WR2B + 512 * 16 * 2;        // 320*320 f16
static constexpr size_t OFF_SAMP = OFF_LB   + 320 * 320 * 2;       // 4096*320 f16
static constexpr size_t OFF_REW  = OFF_SAMP + (size_t)4096 * 320 * 2;  // 4096 f32
static constexpr size_t OFF_MEAN = OFF_REW  + 4096 * 4;            // 320 f32
static constexpr size_t OFF_COV  = OFF_MEAN + 320 * 4;             // 320*320 f32
static constexpr size_t OFF_L    = OFF_COV  + 320 * 320 * 4;       // 320*320 f32
static constexpr size_t OFF_TIDX = OFF_L    + 320 * 320 * 4;       // 128 i32
static constexpr size_t OFF_TC   = OFF_TIDX + 128 * 4;             // 128*320 f32

// ---------------------------------------------------------------------------
// Pack a f32 matrix into WMMA B-tile layout (f16), with zero padding.
// Tiles: 32(K) x 16(N). tile = (k/32)*NT + (n/16).
// In-tile: lane = (n%16) + 16*((k%32)/16) ; half index = k%16.
// trans=1: element (k,n) taken from src[n][k] (used for L^T).
// ---------------------------------------------------------------------------
__global__ void pack_b_kernel(const float* __restrict__ src, _Float16* __restrict__ dst,
                              int K, int N, int R, int C, int trans) {
  int idx = blockIdx.x * blockDim.x + threadIdx.x;
  if (idx >= K * N) return;
  int k = idx / N, n = idx % N;
  float v = 0.0f;
  if (!trans) { if (k < R && n < C) v = src[(size_t)k * C + n]; }
  else        { if (n < R && k < C) v = src[(size_t)n * C + k]; }
  int NT = N >> 4;
  int tile = (k >> 5) * NT + (n >> 4);
  int lanei = (n & 15) + (((k >> 4) & 1) << 4);
  int h = k & 15;
  dst[(size_t)tile * 512 + lanei * 16 + h] = (_Float16)v;
}

__global__ void init_kernel(float* __restrict__ cov, float* __restrict__ L,
                            float* __restrict__ mean) {
  int idx = blockIdx.x * blockDim.x + threadIdx.x;
  if (idx < DDIM * DDIM) {
    int i = idx / DDIM, j = idx % DDIM;
    cov[idx] = (i == j) ? 1.0f : 0.0f;
    L[idx] = 0.0f;
    if (idx < DDIM) mean[idx] = 0.0f;
  }
}

// Single-block Cholesky (320x320). Compute-trivial; done once per CEM iter.
__global__ void cholesky_kernel(const float* __restrict__ A, float* __restrict__ L) {
  int tid = threadIdx.x;
  for (int j = 0; j < DDIM; ++j) {
    if (tid == 0) {
      float s = A[j * DDIM + j];
      for (int k = 0; k < j; ++k) { float v = L[j * DDIM + k]; s -= v * v; }
      L[j * DDIM + j] = sqrtf(fmaxf(s, 1e-12f));
    }
    __syncthreads();
    float dj = L[j * DDIM + j];
    int i = j + 1 + tid;
    if (i < DDIM) {
      float s = A[i * DDIM + j];
      for (int k = 0; k < j; ++k) s -= L[i * DDIM + k] * L[j * DDIM + k];
      L[i * DDIM + j] = s / dj;
    }
    __syncthreads();
  }
}

// ---------------------------------------------------------------------------
// samples = clip(mean + Z @ L^T, -1, 1)  -- WMMA GEMM, Z generated in VGPRs.
// 32 blocks x 256 threads; wave w owns rows [blk*128 + 16w, +16).
// ---------------------------------------------------------------------------
__global__ __launch_bounds__(256) void sample_kernel(
    const _Float16* __restrict__ LB, const float* __restrict__ mean,
    _Float16* __restrict__ samples, int it) {
#if defined(__HIP_DEVICE_COMPILE__)
  int tid = threadIdx.x, w = tid >> 5, lane = tid & 31;
  int row0 = blockIdx.x * 128 + w * 16;
  int khalf = (lane >> 4) << 4;

  AV az[10];  // 10 K-tiles of Z held in registers, reused across 20 N-tiles
#pragma unroll
  for (int ks = 0; ks < 10; ++ks) {
#pragma unroll
    for (int q = 0; q < 8; ++q) {
      int m = row0 + (lane & 15);
      int k0 = 2 * q + khalf + ks * 32;
      unsigned s0 = mix2(mix2(0x51AB7u + (unsigned)it, (unsigned)m), (unsigned)k0);
      float2 nn = rng_normal2(s0);  // pair covers k0, k0+1
      az[ks].h[2 * q]     = (_Float16)nn.x;
      az[ks].h[2 * q + 1] = (_Float16)nn.y;
    }
  }
#pragma unroll 1
  for (int nt = 0; nt < 20; ++nt) {
    v8f acc = {};
#pragma unroll
    for (int ks = 0; ks < 10; ++ks) {
      v16h b = load_b_g(LB, ks * 20 + nt, lane);
      acc = wmma_f16(az[ks].v, b, acc);
    }
    int n = nt * 16 + (lane & 15);
    float mn = mean[n];
#pragma unroll
    for (int q = 0; q < 8; ++q) {
      int m = row0 + q + 8 * (lane >> 4);
      float val = fminf(1.0f, fmaxf(-1.0f, mn + acc[q]));
      samples[(size_t)m * DDIM + n] = (_Float16)val;
    }
  }
#endif
}

// ---------------------------------------------------------------------------
// Fused 20-step stochastic rollout. 32 blocks x 8 waves; wave w owns 16
// candidates (rows [16w,16w+16) of the 128-row block tile). The per-step
// action tile is staged block-wide by the Tensor Data Mover (strided 2D tile
// with LDS pad to hit the 96-elem row pitch); the MLP math is wave-local.
// ---------------------------------------------------------------------------
__global__ __launch_bounds__(256) void rollout_kernel(
    const _Float16* __restrict__ samples, const float* __restrict__ state,
    const _Float16* __restrict__ W1B, const float* __restrict__ b1,
    const _Float16* __restrict__ W2B, const float* __restrict__ b2,
    const _Float16* __restrict__ WR1B, const float* __restrict__ br1,
    const _Float16* __restrict__ WR2B, const float* __restrict__ br2,
    float* __restrict__ rewards, int it) {
#if defined(__HIP_DEVICE_COMPILE__)
  __shared__ _Float16 Xs[128 * 96];   // [row][96] : state(64) | action(16) | pad(16)
  __shared__ _Float16 Hs[128 * 64];   // hidden slice, reused dyn/reward
  __shared__ float    rpart[128 * 2];
  __shared__ float    racc[128];

  const int tid = threadIdx.x;
  const int w = tid >> 5, lane = tid & 31;
  const int cand0 = blockIdx.x * 128;
  const int mrow0 = w * 16;

  if (lane == 0) {
    __builtin_prefetch(W1B, 0, 0);
    __builtin_prefetch(W2B, 0, 0);
    __builtin_prefetch(WR1B, 0, 0);
    __builtin_prefetch(WR2B, 0, 0);
  }

  // init X: broadcast initial state, zero the K-pad columns
  for (int e = lane; e < 16 * 96; e += 32) {
    int r = e / 96, c = e % 96;
    float v = (c < S_DIM) ? state[c] : 0.0f;
    Xs[(mrow0 + r) * 96 + c] = (_Float16)v;
  }
  if (lane < 16) racc[mrow0 + lane] = 0.0f;

  const v8f vzero = {};
  v8f acc2[8];  // out[16,128] accumulators (dyn MLP)
  v8f acc3;     // out[16,16]  accumulators (reward MLP, cols 0..1 real)

#pragma unroll 1
  for (int t = 0; t < H_PLAN; ++t) {
#if HAVE_TDM
    // ---- TDM: stage this block's action tile [128 x 16] into Xs[:,64:80] ----
    __syncthreads();  // all waves done with previous step's X
    if (w == 0) {
      const _Float16* gsrc = samples + (size_t)cand0 * DDIM + t * A_DIM;
      unsigned lds_addr = (unsigned)(uintptr_t)(&Xs[S_DIM]);  // row 0, col 64
      tdm_load_actions(gsrc, lds_addr, 128);
      __builtin_amdgcn_s_wait_tensorcnt(0);
    }
    __syncthreads();  // actions visible block-wide
#else
    for (int e = lane; e < 16 * 16; e += 32) {
      int r = e >> 4, c = e & 15;
      Xs[(mrow0 + r) * 96 + S_DIM + c] =
          samples[(size_t)(cand0 + mrow0 + r) * DDIM + t * A_DIM + c];
    }
#endif
#pragma unroll
    for (int n = 0; n < 8; ++n) acc2[n] = vzero;
    acc3 = vzero;

#pragma unroll 1
    for (int nh = 0; nh < 8; ++nh) {
      // ---- dynamics MLP, layer 1: 64-wide hidden slice ----
      v8f acc1[4] = {vzero, vzero, vzero, vzero};
#pragma unroll
      for (int ks = 0; ks < 3; ++ks) {
        v16h a = load_a_lds(Xs, mrow0, 96, ks * 32, lane);
#pragma unroll
        for (int nt = 0; nt < 4; ++nt) {
          v16h b = load_b_g(W1B, ks * 32 + nh * 4 + nt, lane);  // NT=32
          acc1[nt] = wmma_f16(a, b, acc1[nt]);
        }
      }
#pragma unroll
      for (int nt = 0; nt < 4; ++nt) {
#pragma unroll
        for (int q = 0; q < 8; ++q) {
          int m = mrow0 + q + 8 * (lane >> 4);
          int nl = nt * 16 + (lane & 15);
          float x = acc1[nt][q] + b1[nh * 64 + nl];
          Hs[m * 64 + nl] = (_Float16)fast_tanh(x);
        }
      }
      // ---- dynamics MLP, layer 2 partial-K accumulate ----
#pragma unroll
      for (int ks = 0; ks < 2; ++ks) {
        v16h a = load_a_lds(Hs, mrow0, 64, ks * 32, lane);
#pragma unroll
        for (int nt = 0; nt < 8; ++nt) {
          v16h b = load_b_g(W2B, (nh * 2 + ks) * 8 + nt, lane);  // NT=8
          acc2[nt] = wmma_f16(a, b, acc2[nt]);
        }
      }
      // ---- reward MLP, layer 1 (reuse Hs; wave-local so no barrier) ----
      v8f accr[4] = {vzero, vzero, vzero, vzero};
#pragma unroll
      for (int ks = 0; ks < 3; ++ks) {
        v16h a = load_a_lds(Xs, mrow0, 96, ks * 32, lane);
#pragma unroll
        for (int nt = 0; nt < 4; ++nt) {
          v16h b = load_b_g(WR1B, ks * 32 + nh * 4 + nt, lane);
          accr[nt] = wmma_f16(a, b, accr[nt]);
        }
      }
#pragma unroll
      for (int nt = 0; nt < 4; ++nt) {
#pragma unroll
        for (int q = 0; q < 8; ++q) {
          int m = mrow0 + q + 8 * (lane >> 4);
          int nl = nt * 16 + (lane & 15);
          float x = accr[nt][q] + br1[nh * 64 + nl];
          Hs[m * 64 + nl] = (_Float16)fast_tanh(x);
        }
      }
      // ---- reward MLP, layer 2 (NT=1) ----
#pragma unroll
      for (int ks = 0; ks < 2; ++ks) {
        v16h a = load_a_lds(Hs, mrow0, 64, ks * 32, lane);
        v16h b = load_b_g(WR2B, nh * 2 + ks, lane);
        acc3 = wmma_f16(a, b, acc3);
      }
    }  // nh

    // ---- state update: nxt = mean + softplus(std_pre) * N(0,1) ----
    float2 nn = make_float2(0.0f, 0.0f);
#pragma unroll
    for (int stt = 0; stt < 4; ++stt) {
#pragma unroll
      for (int q = 0; q < 8; ++q) {
        int idx = stt * 8 + q;
        if ((idx & 1) == 0) {
          unsigned seed =
              mix2(mix2(0xA511E9B3u + (unsigned)it, (unsigned)(t * 131 + 7)),
                   (unsigned)(((blockIdx.x * 8 + w) << 9) | (lane << 4) | (idx >> 1)));
          nn = rng_normal2(seed);
        }
        float z = (idx & 1) ? nn.y : nn.x;
        int m = mrow0 + q + 8 * (lane >> 4);
        int s = stt * 16 + (lane & 15);
        float mean_s = acc2[stt][q] + b2[s];
        float stds = softplusf(acc2[4 + stt][q] + b2[S_DIM + s]);
        Xs[m * 96 + s] = (_Float16)(mean_s + stds * z);
      }
    }
    // ---- reward: r = o0 + softplus(o1) * N(0,1) ----
    if ((lane & 15) < 2) {
#pragma unroll
      for (int q = 0; q < 8; ++q) {
        int m = mrow0 + q + 8 * (lane >> 4);
        int n = lane & 15;
        rpart[m * 2 + n] = acc3[q] + br2[n];
      }
    }
    if (lane < 16) {
      int m = mrow0 + lane;
      unsigned seed = mix2(mix2(0xC0FFEEu + (unsigned)it, (unsigned)(t * 977 + 13)),
                           (unsigned)(cand0 + m));
      racc[m] += rpart[m * 2 + 0] + softplusf(rpart[m * 2 + 1]) * rng_normal(seed);
    }
  }  // t

  if (lane < 16) rewards[cand0 + mrow0 + lane] = racc[mrow0 + lane];
#endif
}

// ---------------------------------------------------------------------------
// Top-128 of 4096 rewards: 128 rounds of block-wide argmax extraction
// (descending order, so topidx[0] == argmax for the final answer).
// ---------------------------------------------------------------------------
__global__ void topk_kernel(const float* __restrict__ rewards, int* __restrict__ topidx) {
  __shared__ float vals[N_CAND];
  __shared__ float rmax[256];
  __shared__ int   ridx[256];
  int tid = threadIdx.x;
  for (int i = tid; i < N_CAND; i += 256) vals[i] = rewards[i];
  __syncthreads();
  for (int r = 0; r < TOP_K; ++r) {
    float best = -3.4e38f; int bi = 0;
    for (int i = tid; i < N_CAND; i += 256) {
      float v = vals[i];
      if (v > best) { best = v; bi = i; }
    }
    rmax[tid] = best; ridx[tid] = bi;
    __syncthreads();
    for (int s = 128; s > 0; s >>= 1) {
      if (tid < s && rmax[tid + s] > rmax[tid]) {
        rmax[tid] = rmax[tid + s]; ridx[tid] = ridx[tid + s];
      }
      __syncthreads();
    }
    if (tid == 0) { topidx[r] = ridx[0]; vals[ridx[0]] = -3.4e38f; }
    __syncthreads();
  }
}

// mean over elites + centered matrix Tc (f32) for the covariance update
__global__ void reduce_mean_kernel(const _Float16* __restrict__ samples,
                                   const int* __restrict__ topidx,
                                   float* __restrict__ mean, float* __restrict__ Tc) {
  int d = threadIdx.x;  // 320 threads
  float s = 0.0f;
  for (int i = 0; i < TOP_K; ++i) s += (float)samples[(size_t)topidx[i] * DDIM + d];
  float mu = s * (1.0f / TOP_K);
  mean[d] = mu;
  for (int i = 0; i < TOP_K; ++i)
    Tc[i * DDIM + d] = (float)samples[(size_t)topidx[i] * DDIM + d] - mu;
}

// cov = Tc^T Tc / 127 + COV_REG * I   (rank(cov) <= 128 < 320 always)
__global__ void cov_kernel(const float* __restrict__ Tc, float* __restrict__ cov) {
  int ij = blockIdx.x * blockDim.x + threadIdx.x;
  if (ij >= DDIM * DDIM) return;
  int i = ij / DDIM, j = ij % DDIM;
  float s = 0.0f;
  for (int k = 0; k < TOP_K; ++k) s += Tc[k * DDIM + i] * Tc[k * DDIM + j];
  cov[ij] = s * (1.0f / (TOP_K - 1)) + ((i == j) ? 1e-4f : 0.0f);
}

__global__ void final_kernel(const _Float16* __restrict__ samples,
                             const int* __restrict__ topidx, float* __restrict__ out) {
  int tid = threadIdx.x;
  if (tid < A_DIM) out[tid] = (float)samples[(size_t)topidx[0] * DDIM + tid];
}

// ---------------------------------------------------------------------------
// Host driver: full CEM loop (5 opt steps + final), graph-capture safe.
// ---------------------------------------------------------------------------
extern "C" void kernel_launch(void* const* d_in, const int* in_sizes, int n_in,
                              void* d_out, int out_size, void* d_ws, size_t ws_size,
                              hipStream_t stream) {
  const float* state = (const float*)d_in[0];
  const float* W1  = (const float*)d_in[1];
  const float* b1  = (const float*)d_in[2];
  const float* W2  = (const float*)d_in[3];
  const float* b2  = (const float*)d_in[4];
  const float* Wr1 = (const float*)d_in[5];
  const float* br1 = (const float*)d_in[6];
  const float* Wr2 = (const float*)d_in[7];
  const float* br2 = (const float*)d_in[8];

  char* ws = (char*)d_ws;
  _Float16* W1B  = (_Float16*)(ws + OFF_W1B);
  _Float16* WR1B = (_Float16*)(ws + OFF_WR1B);
  _Float16* W2B  = (_Float16*)(ws + OFF_W2B);
  _Float16* WR2B = (_Float16*)(ws + OFF_WR2B);
  _Float16* LB   = (_Float16*)(ws + OFF_LB);
  _Float16* SAMP = (_Float16*)(ws + OFF_SAMP);
  float* REW  = (float*)(ws + OFF_REW);
  float* MEAN = (float*)(ws + OFF_MEAN);
  float* COV  = (float*)(ws + OFF_COV);
  float* L    = (float*)(ws + OFF_L);
  int*   TIDX = (int*)(ws + OFF_TIDX);
  float* TC   = (float*)(ws + OFF_TC);

  // Pack weights into WMMA B-tile layout (once per launch, deterministic)
  pack_b_kernel<<<(96 * 512 + 255) / 256, 256, 0, stream>>>(W1, W1B, 96, 512, 80, 512, 0);
  pack_b_kernel<<<(96 * 512 + 255) / 256, 256, 0, stream>>>(Wr1, WR1B, 96, 512, 80, 512, 0);
  pack_b_kernel<<<(512 * 128 + 255) / 256, 256, 0, stream>>>(W2, W2B, 512, 128, 512, 128, 0);
  pack_b_kernel<<<(512 * 16 + 255) / 256, 256, 0, stream>>>(Wr2, WR2B, 512, 16, 512, 2, 0);
  init_kernel<<<(DDIM * DDIM + 255) / 256, 256, 0, stream>>>(COV, L, MEAN);

  for (int it = 0; it < 6; ++it) {
    cholesky_kernel<<<1, DDIM, 0, stream>>>(COV, L);
    pack_b_kernel<<<(DDIM * DDIM + 255) / 256, 256, 0, stream>>>(L, LB, DDIM, DDIM,
                                                                 DDIM, DDIM, 1);
    sample_kernel<<<32, 256, 0, stream>>>(LB, MEAN, SAMP, it);
    rollout_kernel<<<32, 256, 0, stream>>>(SAMP, state, W1B, b1, W2B, b2, WR1B, br1,
                                           WR2B, br2, REW, it);
    topk_kernel<<<1, 256, 0, stream>>>(REW, TIDX);
    if (it < 5) {
      reduce_mean_kernel<<<1, DDIM, 0, stream>>>(SAMP, TIDX, MEAN, TC);
      cov_kernel<<<(DDIM * DDIM + 255) / 256, 256, 0, stream>>>(TC, COV);
    }
  }
  final_kernel<<<1, 32, 0, stream>>>(SAMP, TIDX, (float*)d_out);
}
